// GNNLayer_86732569575637
// MI455X (gfx1250) — matrix operand order repaired and checked by reading, not compiled, not run on hardware
//
#include <hip/hip_runtime.h>

// GCNConv + ReLU for MI455X (gfx1250, wave32, WMMA).
// Pipeline:
//   1) deg[i]=1 (self loop); atomicAdd over dst edges; dis=rsqrt(deg)
//   2) WMMA GEMM: h'[i] = (x[i] @ W) * dis[i]   (f16 inputs, f32 accum)
//      -> written to workspace h' AND to d_out (seeds the self-loop term)
//   3) scatter: d_out[dst] += h'[src]  (one wave32 per edge, float4 atomics)
//   4) finalize in place: d_out[i] = relu(dis[i]*d_out[i] + b)

#define D 128

typedef __attribute__((ext_vector_type(16))) _Float16 v16h;
typedef __attribute__((ext_vector_type(8)))  float    v8f;

__global__ void k_init_deg(float* __restrict__ deg, int N) {
    int i = blockIdx.x * blockDim.x + threadIdx.x;
    if (i < N) deg[i] = 1.0f;  // self-loop
}

__global__ void k_count_deg(const int* __restrict__ dst, float* __restrict__ deg, int E) {
    int e = blockIdx.x * blockDim.x + threadIdx.x;
    if (e < E) atomicAdd(&deg[dst[e]], 1.0f);
}

__global__ void k_rsqrt(float* __restrict__ deg, int N) {
    int i = blockIdx.x * blockDim.x + threadIdx.x;
    if (i < N) deg[i] = rsqrtf(deg[i]);
}

// Block: 256 threads = 8 waves. Block tile: 128 rows x 128 cols.
// Wave tile: 16 rows x 128 cols (8 x 16x16 WMMA tiles, K=128 in 4 chunks of 32).
__launch_bounds__(256)
__global__ void k_gemm_scaled(const float* __restrict__ x,
                              const float* __restrict__ W,
                              const float* __restrict__ dis,
                              float* __restrict__ hprime,
                              float* __restrict__ out_acc,
                              int N) {
    // W transposed ([n][k]) in f16, row stride 132 halves (264B) to spread LDS banks.
    __shared__ _Float16 Wt[D * 132];

    for (int idx = threadIdx.x; idx < D * D; idx += blockDim.x) {
        int k = idx >> 7;    // row of W
        int n = idx & 127;   // col of W
        Wt[n * 132 + k] = (_Float16)W[idx];
    }
    __syncthreads();

    const int lane  = threadIdx.x & 31;
    const int wave  = threadIdx.x >> 5;
    const int hi    = (lane >> 4) & 1;   // lane-half select
    const int m     = lane & 15;
    const int rbase = blockIdx.x * 128 + wave * 16;

    // ---- A fragments: 16x32 f16 per K-chunk; ISA layout:
    // lane m holds K = {hi*8 + 0..7} in halves 0..7 and K = {16 + hi*8 + 0..7} in halves 8..15
    const int arow   = rbase + m;
    const int arow_c = arow < N ? arow : (N - 1);   // clamp: keep EXEC all-1 for WMMA
    const float* xr  = x + (size_t)arow_c * D;

    v16h a[4];
#pragma unroll
    for (int c = 0; c < 4; ++c) {
        const float* p = xr + c * 32 + hi * 8;
#pragma unroll
        for (int j = 0; j < 8; ++j) {
            a[c][j]     = (_Float16)p[j];
            a[c][8 + j] = (_Float16)p[16 + j];
        }
    }

    // Per-output-row scale dis[row]: D-matrix VGPR r -> row rbase + r + hi*8
    float sc[8];
    int   srow[8];
#pragma unroll
    for (int r = 0; r < 8; ++r) {
        srow[r] = rbase + r + hi * 8;
        int rc  = srow[r] < N ? srow[r] : 0;
        sc[r]   = dis[rc];
    }

#pragma unroll
    for (int nt = 0; nt < 8; ++nt) {
        v8f acc = {};
#pragma unroll
        for (int c = 0; c < 4; ++c) {
            // B fragment 32x16: lane n = m holds K = {hi*8 + 0..7} in halves 0..7 and K = {16 + hi*8 + 0..7} in halves 8..15
            v16h bfrag;
            const _Float16* wp = &Wt[(nt * 16 + m) * 132 + c * 32 + hi * 8];
#pragma unroll
            for (int j = 0; j < 8; ++j) { bfrag[j] = wp[j]; bfrag[8 + j] = wp[16 + j]; }

            acc = __builtin_amdgcn_wmma_f32_16x16x32_f16(
                /*neg_a=*/false, a[c], /*neg_b=*/false, bfrag,
                /*c_mod=*/(short)0, acc, /*reuse_a=*/false, /*reuse_b=*/false);
        }
        const int col = nt * 16 + m;
#pragma unroll
        for (int r = 0; r < 8; ++r) {
            if (srow[r] < N) {
                float v = acc[r] * sc[r];
                size_t off = (size_t)srow[r] * D + col;
                hprime[off]  = v;
                out_acc[off] = v;   // seeds self-loop contribution
            }
        }
    }
}

// One wave32 per edge: lane l loads h'[src][4l..4l+3] (float4) and atomically
// adds into out[dst][4l..4l+3]. 512 consecutive bytes per wave -> coalesced.
__launch_bounds__(256)
__global__ void k_scatter(const int* __restrict__ src, const int* __restrict__ dst,
                          const float* __restrict__ h, float* __restrict__ out, int E) {
    int gid  = blockIdx.x * blockDim.x + threadIdx.x;
    int e    = gid >> 5;
    int lane = threadIdx.x & 31;
    if (e >= E) return;
    int s = src[e];
    int d = dst[e];
    const float4* hp = (const float4*)(h + (size_t)s * D);
    float4 v = hp[lane];
    float* op = out + (size_t)d * D + lane * 4;
    atomicAdd(op + 0, v.x);
    atomicAdd(op + 1, v.y);
    atomicAdd(op + 2, v.z);
    atomicAdd(op + 3, v.w);
}

__global__ void k_finalize(float* __restrict__ out, const float* __restrict__ dis,
                           const float* __restrict__ bias, int N) {
    int gid   = blockIdx.x * blockDim.x + threadIdx.x;  // one float4 per thread
    int total = N * (D / 4);
    if (gid >= total) return;
    int i = gid >> 5;   // node
    int q = gid & 31;   // quad within row
    float s   = dis[i];
    float4 v  = ((const float4*)out)[gid];
    float4 bb = ((const float4*)bias)[q];
    v.x = fmaxf(fmaf(v.x, s, bb.x), 0.0f);
    v.y = fmaxf(fmaf(v.y, s, bb.y), 0.0f);
    v.z = fmaxf(fmaf(v.z, s, bb.z), 0.0f);
    v.w = fmaxf(fmaf(v.w, s, bb.w), 0.0f);
    ((float4*)out)[gid] = v;
}

extern "C" void kernel_launch(void* const* d_in, const int* in_sizes, int n_in,
                              void* d_out, int out_size, void* d_ws, size_t ws_size,
                              hipStream_t stream) {
    const float* x    = (const float*)d_in[0];
    const int*   ei   = (const int*)d_in[1];
    const float* W    = (const float*)d_in[2];
    const float* bias = (const float*)d_in[3];

    const int N = in_sizes[0] / D;
    const int E = in_sizes[1] / 2;
    const int* src = ei;        // edge_index[0]
    const int* dst = ei + E;    // edge_index[1]

    // workspace: [dis: N floats (rounded to 256)] [h': N*D floats]
    float* dis = (float*)d_ws;
    size_t disElems = ((size_t)N + 255) & ~(size_t)255;
    float* hprime = dis + disElems;

    float* out = (float*)d_out;

    k_init_deg <<<(N + 255) / 256, 256, 0, stream>>>(dis, N);
    k_count_deg<<<(E + 255) / 256, 256, 0, stream>>>(dst, dis, E);
    k_rsqrt    <<<(N + 255) / 256, 256, 0, stream>>>(dis, N);

    k_gemm_scaled<<<(N + 127) / 128, 256, 0, stream>>>(x, W, dis, hprime, out, N);

    // one wave (32 threads) per edge
    long long scatterThreads = (long long)E * 32;
    k_scatter<<<(unsigned)((scatterThreads + 255) / 256), 256, 0, stream>>>(src, dst, hprime, out, E);

    int quads = N * (D / 4);
    k_finalize<<<(quads + 255) / 256, 256, 0, stream>>>(out, dis, bias, N);
}
